// HungarianMatcher_64716567216464
// MI455X (gfx1250) — matrix operand order repaired and checked by reading, not compiled, not run on hardware
//
#include <hip/hip_runtime.h>
#include <hip/hip_bf16.h>

// ---------------------------------------------------------------------------
// HungarianMatcher for gfx1250 (MI455X).
// Heavy phase: two batched [100 x 65536] x [65536 x 32] reductions done with
// v_wmma_f32_16x16x32_bf16, streaming pred_masks once per target tile (memory bound:
// ~0.35 GB -> ~15us at 23.3 TB/s; FLOPs are negligible at WMMA rates).
// ---------------------------------------------------------------------------

typedef __attribute__((ext_vector_type(16))) __bf16 v16bf;
typedef __attribute__((ext_vector_type(8)))  float  v8f;

constexpr int NB = 8;      // batch
constexpr int NQ = 100;    // queries (pred masks)
constexpr int NT = 32;     // targets
constexpr int NC = 81;     // classes incl. no-object
constexpr int HD = 256;    // pred mask H=W
constexpr int HWP = HD * HD;             // 65536
constexpr int NQTILES = (NQ + 15) / 16;  // 7
constexpr int KCHUNKS = 32;
constexpr int KCHUNK = HWP / KCHUNKS;    // 2048

// --------------------------- zero scratch ----------------------------------
__global__ void k_zero(float* __restrict__ p, int n) {
  int i = blockIdx.x * blockDim.x + threadIdx.x;
  if (i < n) p[i] = 0.0f;
}

// ------------------- nearest downsample 512->256 + tm sums -----------------
// ri = (i*512)//256 = 2i, so just stride-2 gather. Output bf16 (0/1 exact).
__global__ void k_downsample(const float* __restrict__ tgt,
                             __bf16* __restrict__ tmds,
                             float* __restrict__ tmsum) {
  const int bt = blockIdx.x;                     // 0..255 = b*32 + t
  const float* src = tgt + (size_t)bt * 512 * 512;
  __bf16* dst = tmds + (size_t)bt * HWP;
  float cnt = 0.0f;
  for (int k = threadIdx.x; k < HWP; k += blockDim.x) {
    int y = k >> 8, x = k & 255;
    float v = src[(size_t)(y * 2) * 512 + (x * 2)];
    dst[k] = (__bf16)v;
    cnt += v;
  }
  for (int o = 16; o > 0; o >>= 1) cnt += __shfl_xor(cnt, o);
  __shared__ float red[8];
  int lane = threadIdx.x & 31, wv = threadIdx.x >> 5;
  if (lane == 0) red[wv] = cnt;
  __syncthreads();
  if (threadIdx.x == 0) {
    float s = 0.0f;
    for (int i = 0; i < (int)(blockDim.x >> 5); i++) s += red[i];
    tmsum[bt] = s;
  }
}

// --------------------------- WMMA GEMM pass --------------------------------
// One wave32 per (16x16 tile, K-chunk). A = pm rows (f32 -> bf16 on the fly),
// second A fragment = sigmoid(pm). B = downsampled target masks (bf16).
// STATS (== ttile 0) waves also fold in the softplus/sigmoid row sums so each
// pm element is touched exactly once. TTILE is a template constant so the
// inner loop has no runtime branches around the WMMAs (EXEC stays all-1s).
template <int TTILE, bool STATS>
__global__ void __launch_bounds__(32)
k_wmma_cost(const float* __restrict__ pm, const __bf16* __restrict__ tmds,
            float* __restrict__ dot_pm, float* __restrict__ dot_ps,
            float* __restrict__ ssum, float* __restrict__ gsum) {
  const int lane  = threadIdx.x;   // 0..31
  const int hf    = lane >> 4;     // half-wave select (K interleave)
  const int r     = lane & 15;
  const int ntile = blockIdx.x;    // 0..6
  const int kc    = blockIdx.y;
  const int b     = blockIdx.z;

  const int  row    = ntile * 16 + r;          // A row this lane streams
  const bool rowok  = row < NQ;
  const int  rclamp = rowok ? row : (NQ - 1);  // keep loads in-bounds
  const int  tcol   = TTILE * 16 + r;          // B column this lane streams

  const float   scale = rowok ? 1.0f : 0.0f;   // zero padded rows, no branches
  const float*  arow  = pm   + ((size_t)(b * NQ + rclamp)) * HWP + kc * KCHUNK;
  const __bf16* brow  = tmds + ((size_t)(b * NT + tcol))   * HWP + kc * KCHUNK;

  v8f acc_pm = {0, 0, 0, 0, 0, 0, 0, 0};
  v8f acc_ps = {0, 0, 0, 0, 0, 0, 0, 0};
  float sp_acc = 0.0f, sg_acc = 0.0f;

  for (int kk = 0; kk < KCHUNK; kk += 32) {
    // 16-bit A 16x32 layout: lanes 0-15 carry K 0..7 / 16..23 of their row,
    // lanes 16-31 carry K 8..15 / 24..31  (ISA 7.12.2).
    const float* a0 = arow + kk + hf * 8;
    float4 f0 = *(const float4*)(a0);
    float4 f1 = *(const float4*)(a0 + 4);
    float4 f2 = *(const float4*)(a0 + 16);
    float4 f3 = *(const float4*)(a0 + 20);
    __builtin_prefetch(a0 + 32, 0, 3);  // speculative global_prefetch_b8

    float av[16] = {f0.x, f0.y, f0.z, f0.w, f1.x, f1.y, f1.z, f1.w,
                    f2.x, f2.y, f2.z, f2.w, f3.x, f3.y, f3.z, f3.w};
    v16bf a_pm, a_ps;
#pragma unroll
    for (int e = 0; e < 16; e++) {
      float x  = av[e] * scale;
      float e0 = __expf(-fabsf(x));                    // v_exp_f32
      float rc = __builtin_amdgcn_rcpf(1.0f + e0);     // v_rcp_f32 (no IEEE div)
      float s  = (x >= 0.0f) ? rc : e0 * rc;           // sigmoid(x)
      a_pm[e] = (__bf16)x;
      a_ps[e] = (__bf16)s;
      if (STATS) {
        sp_acc += fmaxf(x, 0.0f) + __logf(1.0f + e0);  // softplus via v_log_f32
        sg_acc += s;
      }
    }

    // 16-bit B 32x16: lanes 0-15 carry K 0..7 / 16..23 of their column, lanes 16-31
    // carry K 8..15 / 24..31 (same order as A) -> two 16B bf16 loads per lane.
    const __bf16* b0 = brow + kk + hf * 8;
    v16bf bb;
#pragma unroll
    for (int s = 0; s < 8; s++) { bb[s] = b0[s]; bb[s + 8] = b0[s + 16]; }

    acc_pm = __builtin_amdgcn_wmma_f32_16x16x32_bf16(false, a_pm, false, bb,
                                                     (short)0, acc_pm, false, false);
    acc_ps = __builtin_amdgcn_wmma_f32_16x16x32_bf16(false, a_ps, false, bb,
                                                     (short)0, acc_ps, false, false);
  }

  // C/D layout: element j of lane -> C[M = j + hf*8][Ncol = r]
#pragma unroll
  for (int j = 0; j < 8; j++) {
    int orow = ntile * 16 + j + hf * 8;
    if (orow < NQ) {
      size_t o = ((size_t)(b * NQ + orow)) * NT + TTILE * 16 + r;
      atomicAdd(&dot_pm[o], acc_pm[j]);
      atomicAdd(&dot_ps[o], acc_ps[j]);
    }
  }
  if (STATS && rowok) {
    atomicAdd(&ssum[b * NQ + row], sp_acc);
    atomicAdd(&gsum[b * NQ + row], sg_acc);
  }
}

// ----------------------- assemble cost matrix ------------------------------
// One wave per (b,n): softmax over 81 logits via shuffles, lane t emits
// cost[b,n,t] = class + mask + dice, with nan_to_num clamping.
__global__ void __launch_bounds__(32)
k_assemble(const float* __restrict__ logits, const int* __restrict__ labels,
           const float* __restrict__ dot_pm, const float* __restrict__ dot_ps,
           const float* __restrict__ ssum, const float* __restrict__ gsum,
           const float* __restrict__ tmsum, float* __restrict__ cost) {
  const int bn = blockIdx.x;          // 0..799
  const int b = bn / NQ;
  const int lane = threadIdx.x;       // 0..31
  const float* lg = logits + (size_t)bn * NC;

  float mx = -1e30f;
  for (int c = lane; c < NC; c += 32) mx = fmaxf(mx, lg[c]);
  for (int o = 16; o > 0; o >>= 1) mx = fmaxf(mx, __shfl_xor(mx, o));
  float sum = 0.0f;
  for (int c = lane; c < NC; c += 32) sum += __expf(lg[c] - mx);
  for (int o = 16; o > 0; o >>= 1) sum += __shfl_xor(sum, o);
  const float inv = __builtin_amdgcn_rcpf(sum);

  const int t = lane;                 // NT == 32 == wave width
  const int lab = labels[b * NT + t];
  const float cclass = -__expf(lg[lab] - mx) * inv;

  const size_t o = (size_t)bn * NT + t;
  const float invHW = 1.0f / (float)HWP;
  const float cmask = ssum[bn] * invHW - dot_pm[o] * invHW;
  const float cdice = 1.0f - (2.0f * dot_ps[o] + 1.0f) /
                             (gsum[bn] + tmsum[b * NT + t] + 1.0f);
  float c = cclass + cmask + cdice;
  if (c != c) c = 1.0e6f;             // nan
  c = fminf(fmaxf(c, -1.0e6f), 1.0e6f);
  cost[o] = c;
}

// --------------------- Hungarian (JV) per batch ----------------------------
// N=100 > T=32 so solve on the transposed [32 x 100] matrix like the
// reference, then swap back. Serial on thread 0, matrix staged in LDS.
__global__ void k_hungarian(const float* __restrict__ cost, float* __restrict__ out) {
  __shared__ float  Ct[NT][NQ];          // Ct[target][query]
  __shared__ double u[NT + 1], v[NQ + 1], minv[NQ + 1];
  __shared__ int    p[NQ + 1], way[NQ + 1];
  __shared__ bool   used[NQ + 1];
  const int b = blockIdx.x;

  for (int i = threadIdx.x; i < NQ * NT; i += blockDim.x) {
    int n = i / NT, t = i % NT;
    Ct[t][n] = cost[((size_t)b * NQ + n) * NT + t];
  }
  __syncthreads();

  if (threadIdx.x == 0) {
    for (int j = 0; j <= NQ; j++) { p[j] = 0; v[j] = 0.0; way[j] = 0; }
    for (int i = 0; i <= NT; i++) u[i] = 0.0;
    for (int i = 1; i <= NT; i++) {
      p[0] = i;
      int j0 = 0;
      for (int j = 0; j <= NQ; j++) { minv[j] = 1e30; used[j] = false; }
      while (true) {
        used[j0] = true;
        int i0 = p[j0];
        double delta = 1e30;
        int j1 = 0;
        for (int j = 1; j <= NQ; j++) {
          if (!used[j]) {
            double cur = (double)Ct[i0 - 1][j - 1] - u[i0] - v[j];
            if (cur < minv[j]) { minv[j] = cur; way[j] = j0; }
            if (minv[j] < delta) { delta = minv[j]; j1 = j; }
          }
        }
        for (int j = 0; j <= NQ; j++) {
          if (used[j]) { u[p[j]] += delta; v[j] -= delta; }
          else         { minv[j] -= delta; }
        }
        j0 = j1;
        if (p[j0] == 0) break;
      }
      while (j0) { int j1 = way[j0]; p[j0] = p[j1]; j0 = j1; }
    }
    // Emit pairs sorted by query index ascending (j ascending does this).
    int idx = 0;
    for (int j = 1; j <= NQ; j++) {
      if (p[j] != 0) {
        out[b * NT + idx]           = (float)(j - 1);     // src (query)
        out[NB * NT + b * NT + idx] = (float)(p[j] - 1);  // tgt (target)
        idx++;
      }
    }
  }
}

// ---------------------------------------------------------------------------
extern "C" void kernel_launch(void* const* d_in, const int* in_sizes, int n_in,
                              void* d_out, int out_size, void* d_ws, size_t ws_size,
                              hipStream_t stream) {
  const float* pred_logits   = (const float*)d_in[0];  // [8,100,81]
  const float* pred_masks    = (const float*)d_in[1];  // [8,100,256,256]
  const float* target_masks  = (const float*)d_in[2];  // [8,32,512,512]
  const int*   target_labels = (const int*)d_in[3];    // [8,32]
  float* out = (float*)d_out;                          // [2,8,32] (src, tgt)

  // Workspace layout
  char* ws = (char*)d_ws;
  __bf16* tmds = (__bf16*)ws;                                // 32 MiB bf16
  size_t off = (size_t)NB * NT * HWP * sizeof(__bf16);       // 33,554,432
  float* dot_pm = (float*)(ws + off); off += (size_t)NB * NQ * NT * 4;
  float* dot_ps = (float*)(ws + off); off += (size_t)NB * NQ * NT * 4;
  float* ssum   = (float*)(ws + off); off += (size_t)NB * NQ * 4;
  float* gsum   = (float*)(ws + off); off += (size_t)NB * NQ * 4;
  float* tmsum  = (float*)(ws + off); off += (size_t)NB * NT * 4;
  float* cost   = (float*)(ws + off);

  // 1) zero the atomic accumulators (dot_pm..tmsum are contiguous)
  const int nzero = 2 * NB * NQ * NT + 2 * NB * NQ + NB * NT;  // 53,056
  k_zero<<<(nzero + 255) / 256, 256, 0, stream>>>(dot_pm, nzero);

  // 2) compact + downsample target masks to bf16, tm row sums
  k_downsample<<<NB * NT, 256, 0, stream>>>(target_masks, tmds, tmsum);

  // 3) WMMA streaming pass: dot products (+ row sums on ttile-0 waves)
  dim3 grid(NQTILES, KCHUNKS, NB);   // 7 x 32 x 8 waves per ttile
  k_wmma_cost<0, true ><<<grid, 32, 0, stream>>>(pred_masks, tmds, dot_pm, dot_ps, ssum, gsum);
  k_wmma_cost<1, false><<<grid, 32, 0, stream>>>(pred_masks, tmds, dot_pm, dot_ps, ssum, gsum);

  // 4) assemble cost matrix (class softmax + mask + dice, nan_to_num)
  k_assemble<<<NB * NQ, 32, 0, stream>>>(pred_logits, target_labels,
                                         dot_pm, dot_ps, ssum, gsum, tmsum, cost);

  // 5) Hungarian assignment per batch
  k_hungarian<<<NB, 128, 0, stream>>>(cost, out);
}
